// BioGuardGAT_25537875542439
// MI455X (gfx1250) — matrix operand order repaired and checked by reading, not compiled, not run on hardware
//
#include <hip/hip_runtime.h>
#include <hip/hip_bf16.h>
#include <math.h>

// ---------------- problem constants (match reference) ----------------
#define NNODES   65536
#define NEDGES   131072
#define NGRAPHS  2048
#define NODE_DIM 41
#define EDGE_DIM 6
#define EMB      128
#define HEADS    4
#define E2       (NEDGES + NNODES)   // edges + self loops = 196608
#define NEG_SLOPE 0.2f
#define NEG_INF  (-3.0e38f)

typedef __attribute__((ext_vector_type(16))) _Float16 v16h;
typedef __attribute__((ext_vector_type(8)))  _Float16 v8h;
typedef __attribute__((ext_vector_type(8)))  float    v8f;

// ---------------- helpers ----------------
__device__ __forceinline__ void atomicMaxF(float* addr, float v) {
  unsigned int* ia = (unsigned int*)addr;
  unsigned int cur = __float_as_uint(*addr);
  while (__uint_as_float(cur) < v) {
    unsigned int prev = atomicCAS(ia, cur, __float_as_uint(v));
    if (prev == cur) break;
    cur = prev;
  }
}

__global__ void fill_kernel(float* __restrict__ p, long long n, float v) {
  long long i = (long long)blockIdx.x * blockDim.x + threadIdx.x;
  if (i < n) p[i] = v;
}

// =====================================================================
// LDS-staged WMMA GEMM: C[MxN] = A[MxK] @ B[KxN], row-major fp32 in/out.
// Block = 256 threads (8 waves) -> one 32x128 output strip (2 M-tiles /
// wave sharing one B fragment -> 2 x v_wmma per staged chunk).
// Branch-free staging: addresses are clamped (always valid), A values at
// k>=K are zeroed with a cndmask select; B out-of-range values are finite
// garbage that is either multiplied by A-zeros or read only by inactive
// waves. Loads batch into clauses with a single wait.
// Requirements: M % 32 == 0, N % 4 == 0 (true for all call sites).
// act: 0 = none, 1 = relu. bias may be null.
// =====================================================================
__global__ __launch_bounds__(256) void wmma_gemm_f16(
    const float* __restrict__ A, const float* __restrict__ B,
    float* __restrict__ C, int M, int N, int K,
    const float* __restrict__ bias, int act) {
  // rows padded to 40 halves (80 B): 16-B aligned, spreads LDS banks
  __shared__ __align__(16) _Float16 As[32][40];     // As[m][k]
  __shared__ __align__(16) _Float16 BsT[128][40];   // BsT[n][k]

  int tilesN  = N >> 4;
  int stripsN = (tilesN + 7) >> 3;
  int tm  = blockIdx.x / stripsN;          // 32-row strip index
  int tn0 = (blockIdx.x % stripsN) << 3;   // first 16-col tile of the strip
  int n0  = tn0 << 4;                      // first column of the strip

  int t = threadIdx.x;
  int wid = t >> 5, lane = t & 31;
  int r = lane & 15, g = lane >> 4;
  int tn = tn0 + wid;
  bool active = (tn < tilesN);             // wave-uniform

  const float* Abase = A + (size_t)tm * 32 * K;

  v8f acc0 = {}, acc1 = {};
  for (int k0 = 0; k0 < K; k0 += 32) {
    __syncthreads();                       // protect previous chunk's reads

    // ---- stage A: 32x32, 4 scalar loads / thread, coalesced along k ----
#pragma unroll
    for (int u = 0; u < 4; ++u) {
      int i = t + (u << 8);
      int row = i >> 5, k = i & 31;
      int kk = k0 + k;
      int kc = (kk < K) ? kk : (K - 1);    // clamped, always valid
      float v = Abase[(size_t)row * K + kc];
      if (kk >= K) v = 0.0f;               // cndmask, no branch
      As[row][k] = (_Float16)v;
    }
    // ---- stage B^T: 32x128, 4 x float4 loads / thread, coalesced ----
#pragma unroll
    for (int u = 0; u < 4; ++u) {
      int i = (t << 2) + (u << 10);
      int row = i >> 7, n = i & 127;
      int kk = k0 + row;
      int kc = (kk < K) ? kk : (K - 1);            // row clamp (A zeros cover it)
      int nn = n0 + n;
      int nc = (nn <= N - 4) ? nn : (N - 4);       // col clamp (inactive waves only)
      float4 v4 = *(const float4*)&B[(size_t)kc * N + nc];
      BsT[n + 0][row] = (_Float16)v4.x;
      BsT[n + 1][row] = (_Float16)v4.y;
      BsT[n + 2][row] = (_Float16)v4.z;
      BsT[n + 3][row] = (_Float16)v4.w;
    }
    __syncthreads();

    if (active) {
      // B fragment (shared by both M-tiles): j<8 -> k = 8g + j ; j>=8 -> k = 16 + 8g + (j-8), col = wid*16 + r
      int colL = (wid << 4) + r;
      v8h blo = *(const v8h*)&BsT[colL][g << 3];
      v8h bhi = *(const v8h*)&BsT[colL][16 + (g << 3)];
      v16h b = __builtin_shufflevector(blo, bhi, 0,1,2,3,4,5,6,7,8,9,10,11,12,13,14,15);
      // A fragments: j<8 -> k = 8g + j ; j>=8 -> k = 16 + 8g + (j-8)
      v8h a0lo = *(const v8h*)&As[r][g << 3];
      v8h a0hi = *(const v8h*)&As[r][16 + (g << 3)];
      v8h a1lo = *(const v8h*)&As[16 + r][g << 3];
      v8h a1hi = *(const v8h*)&As[16 + r][16 + (g << 3)];
      v16h a0 = __builtin_shufflevector(a0lo, a0hi, 0,1,2,3,4,5,6,7,8,9,10,11,12,13,14,15);
      v16h a1 = __builtin_shufflevector(a1lo, a1hi, 0,1,2,3,4,5,6,7,8,9,10,11,12,13,14,15);
      acc0 = __builtin_amdgcn_wmma_f32_16x16x32_f16(false, a0, false, b, (short)0, acc0, false, false);
      acc1 = __builtin_amdgcn_wmma_f32_16x16x32_f16(false, a1, false, b, (short)0, acc1, false, false);
    }
  }

  if (active) {
    int n = (tn << 4) + r;
    float bv = bias ? bias[n] : 0.0f;
#pragma unroll
    for (int v = 0; v < 8; ++v) {
      int ml = (g << 3) + v;               // C/D layout: VGPR v -> M = v + 8g
      int m0 = tm * 32 + ml;
      int m1 = tm * 32 + 16 + ml;
      float o0 = acc0[v] + bv;
      float o1 = acc1[v] + bv;
      if (act == 1) { o0 = fmaxf(o0, 0.0f); o1 = fmaxf(o1, 0.0f); }
      C[(size_t)m0 * N + n] = o0;
      C[(size_t)m1 * N + n] = o1;
    }
  }
}

// ---------------- degree + self-loop attr (fill_value='mean') ----------------
__global__ void deg_loop_kernel(const int* __restrict__ ei,
                                const float* __restrict__ eattr,
                                float* __restrict__ deg,
                                float* __restrict__ loopattr) {
  int e = blockIdx.x * blockDim.x + threadIdx.x;
  if (e >= NEDGES) return;
  int d = ei[NEDGES + e];
  atomicAdd(&deg[d], 1.0f);
#pragma unroll
  for (int k = 0; k < EDGE_DIM; ++k)
    atomicAdd(&loopattr[(size_t)d * EDGE_DIM + k], eattr[(size_t)e * EDGE_DIM + k]);
}

__global__ void loop_norm_kernel(float* __restrict__ loopattr, const float* __restrict__ deg) {
  int i = blockIdx.x * blockDim.x + threadIdx.x;
  if (i >= NNODES * EDGE_DIM) return;
  loopattr[i] /= fmaxf(deg[i / EDGE_DIM], 1.0f);
}

// ---------------- per-edge attention score (one wave per (edge, head)) ----------------
__global__ void edge_score_kernel(const int* __restrict__ ei,
                                  const float* __restrict__ eattr,
                                  const float* __restrict__ loopattr,
                                  const float* __restrict__ xl,
                                  const float* __restrict__ xr,
                                  const float* __restrict__ We,   // [EDGE_DIM, H*EMB]
                                  const float* __restrict__ att,  // [H, EMB]
                                  float* __restrict__ score,      // [E2, H]
                                  float* __restrict__ mx,         // [N, H]
                                  int H) {
  int wave = blockIdx.x * (blockDim.x >> 5) + (threadIdx.x >> 5);
  if (wave >= E2 * H) return;
  int e = wave / H, h = wave % H;
  int lane = threadIdx.x & 31;

  int s, d; const float* ea;
  if (e < NEDGES) { s = ei[e]; d = ei[NEDGES + e]; ea = eattr + (size_t)e * EDGE_DIM; }
  else            { s = e - NEDGES; d = s;          ea = loopattr + (size_t)(e - NEDGES) * EDGE_DIM; }

  float eav[EDGE_DIM];
#pragma unroll
  for (int k = 0; k < EDGE_DIM; ++k) eav[k] = ea[k];

  int ld = H * EMB;
  float sc = 0.0f;
#pragma unroll
  for (int i = 0; i < 4; ++i) {
    int c = lane + 32 * i;
    int col = h * EMB + c;
    float ep = 0.0f;
#pragma unroll
    for (int k = 0; k < EDGE_DIM; ++k) ep += eav[k] * We[k * ld + col];
    float m = xl[(size_t)s * ld + col] + xr[(size_t)d * ld + col] + ep;
    float a = (m > 0.0f) ? m : NEG_SLOPE * m;   // LeakyReLU
    sc += a * att[h * EMB + c];
  }
#pragma unroll
  for (int o = 16; o > 0; o >>= 1) sc += __shfl_xor(sc, o, 32);
  if (lane == 0) {
    score[(size_t)e * H + h] = sc;
    atomicMaxF(&mx[(size_t)d * H + h], sc);
  }
}

// ---------------- softmax: exp & denominator (score overwritten in place) ----------------
__global__ void edge_exp_kernel(const int* __restrict__ ei,
                                float* __restrict__ score,
                                const float* __restrict__ mx,
                                float* __restrict__ den, int H) {
  int idx = blockIdx.x * blockDim.x + threadIdx.x;
  if (idx >= E2 * H) return;
  int e = idx / H, h = idx % H;
  int d = (e < NEDGES) ? ei[NEDGES + e] : (e - NEDGES);
  float ex = __expf(score[idx] - mx[(size_t)d * H + h]);
  score[idx] = ex;
  atomicAdd(&den[(size_t)d * H + h], ex);
}

// ---------------- weighted scatter aggregation: out[dst] += alpha * xl[src] ----------------
__global__ void edge_agg_kernel(const int* __restrict__ ei,
                                const float* __restrict__ score,  // exp values
                                const float* __restrict__ den,
                                const float* __restrict__ xl,
                                float* __restrict__ out, int H) {
  int wave = blockIdx.x * (blockDim.x >> 5) + (threadIdx.x >> 5);
  if (wave >= E2 * H) return;
  int e = wave / H, h = wave % H;
  int lane = threadIdx.x & 31;
  int s, d;
  if (e < NEDGES) { s = ei[e]; d = ei[NEDGES + e]; } else { s = d = e - NEDGES; }
  float alpha = score[(size_t)e * H + h] / den[(size_t)d * H + h];
  int ld = H * EMB;
#pragma unroll
  for (int i = 0; i < 4; ++i) {
    int col = h * EMB + lane + 32 * i;
    atomicAdd(&out[(size_t)d * ld + col], xl[(size_t)s * ld + col] * alpha);
  }
}

// ---------------- bias + ELU ----------------
__global__ void bias_elu_kernel(float* __restrict__ h, const float* __restrict__ bias,
                                long long total, int ld) {
  long long idx = (long long)blockIdx.x * blockDim.x + threadIdx.x;
  if (idx >= total) return;
  float v = h[idx] + bias[idx % ld];
  h[idx] = (v > 0.0f) ? v : (__expf(v) - 1.0f);
}

// ---------------- global mean/max pooling ----------------
__global__ void pool_kernel(const float* __restrict__ h, const int* __restrict__ batch,
                            float* __restrict__ cnt, float* __restrict__ pmean,
                            float* __restrict__ pmax) {
  long long idx = (long long)blockIdx.x * blockDim.x + threadIdx.x;
  if (idx >= (long long)NNODES * EMB) return;
  int n = (int)(idx / EMB), c = (int)(idx % EMB);
  int g = batch[n];
  float v = h[idx];
  atomicAdd(&pmean[(size_t)g * EMB + c], v);
  atomicMaxF(&pmax[(size_t)g * EMB + c], v);
  if (c == 0) atomicAdd(&cnt[g], 1.0f);
}

__global__ void pool_fin_kernel(const float* __restrict__ cnt, const float* __restrict__ pmean,
                                const float* __restrict__ pmax, float* __restrict__ vec) {
  int idx = blockIdx.x * blockDim.x + threadIdx.x;
  if (idx >= NGRAPHS * EMB) return;
  int g = idx / EMB, c = idx % EMB;
  float ct = cnt[g];
  vec[(size_t)g * 256 + c]       = pmean[idx] / fmaxf(ct, 1.0f);
  vec[(size_t)g * 256 + EMB + c] = (ct > 0.0f) ? pmax[idx] : 0.0f;
}

// ---------------- siamese combine: [a+b, |a-b|, a*b] ----------------
__global__ void combine_kernel(const float* __restrict__ va, const float* __restrict__ vb,
                               float* __restrict__ comb) {
  int idx = blockIdx.x * blockDim.x + threadIdx.x;
  if (idx >= NGRAPHS * 256) return;
  int g = idx / 256, c = idx % 256;
  float a = va[idx], b = vb[idx];
  comb[(size_t)g * 768 + c]       = a + b;
  comb[(size_t)g * 768 + 256 + c] = fabsf(a - b);
  comb[(size_t)g * 768 + 512 + c] = a * b;
}

// ---------------- BatchNorm (eval: mean=0, var=1) + ReLU ----------------
__global__ void bn_relu_kernel(float* __restrict__ h, const float* __restrict__ gam,
                               const float* __restrict__ bet, long long total, int ld) {
  long long idx = (long long)blockIdx.x * blockDim.x + threadIdx.x;
  if (idx >= total) return;
  int c = (int)(idx % ld);
  float v = gam[c] * h[idx] * rsqrtf(1.0f + 1e-5f) + bet[c];
  h[idx] = fmaxf(v, 0.0f);
}

// ---------------- final head: [G,64] @ [64,1] + b ----------------
__global__ void out_head_kernel(const float* __restrict__ t2, const float* __restrict__ ow,
                                const float* __restrict__ ob, float* __restrict__ out) {
  int wave = blockIdx.x * (blockDim.x >> 5) + (threadIdx.x >> 5);
  if (wave >= NGRAPHS) return;
  int lane = threadIdx.x & 31;
  float s = t2[(size_t)wave * 64 + lane] * ow[lane] +
            t2[(size_t)wave * 64 + 32 + lane] * ow[32 + lane];
#pragma unroll
  for (int o = 16; o > 0; o >>= 1) s += __shfl_xor(s, o, 32);
  if (lane == 0) out[wave] = s + ob[0];
}

// =====================================================================
//                              host side
// =====================================================================
static inline int blocksFor(long long threads, int blk) {
  return (int)((threads + blk - 1) / blk);
}

static inline int gemmBlocks(int M, int N) {
  int tilesM  = M >> 5;                    // 32-row strips
  int stripsN = ((N >> 4) + 7) >> 3;       // 128-col strips
  return tilesM * stripsN;
}

struct ArmWs {
  float *deg, *loopattr, *xl1, *xr1, *h1, *score1, *mx1, *den1;
  float *xl2, *xr2, *h2, *score2, *mx2, *den2;
  float *cnt, *pmean, *pmax;
};

static void run_arm(const float* x, const int* ei, const float* eattr, const int* batch,
                    const float* W1l, const float* W1r, const float* W1e,
                    const float* att1, const float* b1,
                    const float* W2l, const float* W2r, const float* W2e,
                    const float* att2, const float* b2,
                    const ArmWs& w, float* vec, hipStream_t st) {
  const int BK = 256;

  // init accumulators (workspace is poisoned by the harness)
  fill_kernel<<<blocksFor(NNODES, BK), BK, 0, st>>>(w.deg, NNODES, 0.0f);
  fill_kernel<<<blocksFor((long long)NNODES * EDGE_DIM, BK), BK, 0, st>>>(w.loopattr, (long long)NNODES * EDGE_DIM, 0.0f);
  fill_kernel<<<blocksFor((long long)NNODES * HEADS * EMB, BK), BK, 0, st>>>(w.h1, (long long)NNODES * HEADS * EMB, 0.0f);
  fill_kernel<<<blocksFor((long long)NNODES * HEADS, BK), BK, 0, st>>>(w.mx1, (long long)NNODES * HEADS, NEG_INF);
  fill_kernel<<<blocksFor((long long)NNODES * HEADS, BK), BK, 0, st>>>(w.den1, (long long)NNODES * HEADS, 0.0f);
  fill_kernel<<<blocksFor((long long)NNODES * EMB, BK), BK, 0, st>>>(w.h2, (long long)NNODES * EMB, 0.0f);
  fill_kernel<<<blocksFor(NNODES, BK), BK, 0, st>>>(w.mx2, NNODES, NEG_INF);
  fill_kernel<<<blocksFor(NNODES, BK), BK, 0, st>>>(w.den2, NNODES, 0.0f);
  fill_kernel<<<blocksFor(NGRAPHS, BK), BK, 0, st>>>(w.cnt, NGRAPHS, 0.0f);
  fill_kernel<<<blocksFor((long long)NGRAPHS * EMB, BK), BK, 0, st>>>(w.pmean, (long long)NGRAPHS * EMB, 0.0f);
  fill_kernel<<<blocksFor((long long)NGRAPHS * EMB, BK), BK, 0, st>>>(w.pmax, (long long)NGRAPHS * EMB, NEG_INF);

  // self-loop attr = mean of incoming edge_attr
  deg_loop_kernel<<<blocksFor(NEDGES, BK), BK, 0, st>>>(ei, eattr, w.deg, w.loopattr);
  loop_norm_kernel<<<blocksFor((long long)NNODES * EDGE_DIM, BK), BK, 0, st>>>(w.loopattr, w.deg);

  // ---- layer 1: GATv2(41 -> 128, heads=4) ----
  wmma_gemm_f16<<<gemmBlocks(NNODES, HEADS * EMB), BK, 0, st>>>(
      x, W1l, w.xl1, NNODES, HEADS * EMB, NODE_DIM, nullptr, 0);
  wmma_gemm_f16<<<gemmBlocks(NNODES, HEADS * EMB), BK, 0, st>>>(
      x, W1r, w.xr1, NNODES, HEADS * EMB, NODE_DIM, nullptr, 0);

  edge_score_kernel<<<blocksFor((long long)E2 * HEADS * 32, BK), BK, 0, st>>>(
      ei, eattr, w.loopattr, w.xl1, w.xr1, W1e, att1, w.score1, w.mx1, HEADS);
  edge_exp_kernel<<<blocksFor((long long)E2 * HEADS, BK), BK, 0, st>>>(ei, w.score1, w.mx1, w.den1, HEADS);
  edge_agg_kernel<<<blocksFor((long long)E2 * HEADS * 32, BK), BK, 0, st>>>(
      ei, w.score1, w.den1, w.xl1, w.h1, HEADS);
  bias_elu_kernel<<<blocksFor((long long)NNODES * HEADS * EMB, BK), BK, 0, st>>>(
      w.h1, b1, (long long)NNODES * HEADS * EMB, HEADS * EMB);

  // ---- layer 2: GATv2(512 -> 128, heads=1) ----
  wmma_gemm_f16<<<gemmBlocks(NNODES, EMB), BK, 0, st>>>(
      w.h1, W2l, w.xl2, NNODES, EMB, HEADS * EMB, nullptr, 0);
  wmma_gemm_f16<<<gemmBlocks(NNODES, EMB), BK, 0, st>>>(
      w.h1, W2r, w.xr2, NNODES, EMB, HEADS * EMB, nullptr, 0);

  edge_score_kernel<<<blocksFor((long long)E2 * 32, BK), BK, 0, st>>>(
      ei, eattr, w.loopattr, w.xl2, w.xr2, W2e, att2, w.score2, w.mx2, 1);
  edge_exp_kernel<<<blocksFor((long long)E2, BK), BK, 0, st>>>(ei, w.score2, w.mx2, w.den2, 1);
  edge_agg_kernel<<<blocksFor((long long)E2 * 32, BK), BK, 0, st>>>(
      ei, w.score2, w.den2, w.xl2, w.h2, 1);
  bias_elu_kernel<<<blocksFor((long long)NNODES * EMB, BK), BK, 0, st>>>(
      w.h2, b2, (long long)NNODES * EMB, EMB);

  // ---- pooling -> [G, 256] ----
  pool_kernel<<<blocksFor((long long)NNODES * EMB, BK), BK, 0, st>>>(w.h2, batch, w.cnt, w.pmean, w.pmax);
  pool_fin_kernel<<<blocksFor((long long)NGRAPHS * EMB, BK), BK, 0, st>>>(w.cnt, w.pmean, w.pmax, vec);
}

extern "C" void kernel_launch(void* const* d_in, const int* in_sizes, int n_in,
                              void* d_out, int out_size, void* d_ws, size_t ws_size,
                              hipStream_t stream) {
  (void)in_sizes; (void)n_in; (void)out_size; (void)ws_size;
  const int BK = 256;

  const float* x_a   = (const float*)d_in[0];
  const int*   ei_a  = (const int*)d_in[1];
  const float* ea_a  = (const float*)d_in[2];
  const int*   bat_a = (const int*)d_in[3];
  const float* x_b   = (const float*)d_in[4];
  const int*   ei_b  = (const int*)d_in[5];
  const float* ea_b  = (const float*)d_in[6];
  const int*   bat_b = (const int*)d_in[7];
  const float* W1l   = (const float*)d_in[8];
  const float* W1r   = (const float*)d_in[9];
  const float* W1e   = (const float*)d_in[10];
  const float* att1  = (const float*)d_in[11];
  const float* b1    = (const float*)d_in[12];
  const float* W2l   = (const float*)d_in[13];
  const float* W2r   = (const float*)d_in[14];
  const float* W2e   = (const float*)d_in[15];
  const float* att2  = (const float*)d_in[16];
  const float* b2    = (const float*)d_in[17];
  const float* fc1w  = (const float*)d_in[18];
  const float* fc1b  = (const float*)d_in[19];
  const float* bng   = (const float*)d_in[20];
  const float* bnb   = (const float*)d_in[21];
  const float* fc2w  = (const float*)d_in[22];
  const float* fc2b  = (const float*)d_in[23];
  const float* outw  = (const float*)d_in[24];
  const float* outb  = (const float*)d_in[25];

  // ---- workspace layout (floats) ----
  float* ws = (float*)d_ws;
  size_t off = 0;
  ArmWs w;
  w.deg      = ws + off; off += NNODES;
  w.loopattr = ws + off; off += (size_t)NNODES * EDGE_DIM;
  w.xl1      = ws + off; off += (size_t)NNODES * HEADS * EMB;
  w.xr1      = ws + off; off += (size_t)NNODES * HEADS * EMB;
  w.h1       = ws + off; off += (size_t)NNODES * HEADS * EMB;
  w.score1   = ws + off; off += (size_t)E2 * HEADS;
  w.mx1      = ws + off; off += (size_t)NNODES * HEADS;
  w.den1     = ws + off; off += (size_t)NNODES * HEADS;
  w.xl2      = ws + off; off += (size_t)NNODES * EMB;
  w.xr2      = ws + off; off += (size_t)NNODES * EMB;
  w.h2       = ws + off; off += (size_t)NNODES * EMB;
  w.score2   = ws + off; off += (size_t)E2;
  w.mx2      = ws + off; off += NNODES;
  w.den2     = ws + off; off += NNODES;
  w.cnt      = ws + off; off += NGRAPHS;
  w.pmean    = ws + off; off += (size_t)NGRAPHS * EMB;
  w.pmax     = ws + off; off += (size_t)NGRAPHS * EMB;
  float* vec_a = ws + off; off += (size_t)NGRAPHS * 256;
  float* vec_b = ws + off; off += (size_t)NGRAPHS * 256;
  float* comb  = ws + off; off += (size_t)NGRAPHS * 768;
  float* t1    = ws + off; off += (size_t)NGRAPHS * 256;
  float* t2    = ws + off; off += (size_t)NGRAPHS * 64;

  // ---- both arms (sequential on stream; workspace reused) ----
  run_arm(x_a, ei_a, ea_a, bat_a, W1l, W1r, W1e, att1, b1, W2l, W2r, W2e, att2, b2, w, vec_a, stream);
  run_arm(x_b, ei_b, ea_b, bat_b, W1l, W1r, W1e, att1, b1, W2l, W2r, W2e, att2, b2, w, vec_b, stream);

  // ---- classifier ----
  combine_kernel<<<blocksFor((long long)NGRAPHS * 256, BK), BK, 0, stream>>>(vec_a, vec_b, comb);

  // fc1: [2048,768] @ [768,256] + b  -> bn -> relu
  wmma_gemm_f16<<<gemmBlocks(NGRAPHS, 256), BK, 0, stream>>>(comb, fc1w, t1, NGRAPHS, 256, 768, fc1b, 0);
  bn_relu_kernel<<<blocksFor((long long)NGRAPHS * 256, BK), BK, 0, stream>>>(
      t1, bng, bnb, (long long)NGRAPHS * 256, 256);

  // fc2: [2048,256] @ [256,64] + b -> relu (fused epilogue)
  wmma_gemm_f16<<<gemmBlocks(NGRAPHS, 64), BK, 0, stream>>>(t1, fc2w, t2, NGRAPHS, 64, 256, fc2b, 1);

  // out head: [2048,64] @ [64,1] + b
  out_head_kernel<<<blocksFor((long long)NGRAPHS * 32, BK), BK, 0, stream>>>(t2, outw, outb, (float*)d_out);
}
